// SelfAttention_77232101916706
// MI455X (gfx1250) — compile-verified, matrix operand order checked by reading, not run on hardware
//
#include <hip/hip_runtime.h>
#include <hip/hip_bf16.h>
#include <stdint.h>

// ---------------- problem constants ----------------
#define DIMD   1024
#define NHEAD  16
#define HD     64
#define SEQ    2048
#define BATCH  2
#define MROWS  (BATCH * SEQ)   // 4096
#define QKV_N  (3 * DIMD)      // 3072

typedef __attribute__((ext_vector_type(16))) _Float16 v16h;
typedef __attribute__((ext_vector_type(8)))  _Float16 v8h;
typedef __attribute__((ext_vector_type(4)))  _Float16 v4h;
typedef __attribute__((ext_vector_type(8)))  float    v8f;

// ---------------- CDNA5 async global->LDS (ASYNCcnt path) ----------------
static __device__ __forceinline__ void async_ld_b128(void* lds_dst, const void* gsrc) {
    uint32_t lds_off = (uint32_t)(uintptr_t)lds_dst;   // low 32 bits of generic LDS addr
    asm volatile("global_load_async_to_lds_b128 %0, %1, off"
                 :: "v"(lds_off), "v"(gsrc) : "memory");
}
// ASYNCcnt completes in-order: waiting <=N after issuing N new loads guarantees
// the previous tile's loads have landed in LDS.
static __device__ __forceinline__ void async_wait0() {
    asm volatile("s_wait_asynccnt 0" ::: "memory");
}
static __device__ __forceinline__ void async_wait4() {
    asm volatile("s_wait_asynccnt 4" ::: "memory");
}
static __device__ __forceinline__ void async_wait8() {
    asm volatile("s_wait_asynccnt 8" ::: "memory");
}

// ---------------- 16x32 f16 fragment load (A layout; mirror used for B) ----
// 16-bit A 16x32 layout: lanes 0-15 hold M=lane, K chunks [0..7],[16..23];
// lanes 16-31 hold M=lane-16, K chunks [8..15],[24..31].
static __device__ __forceinline__ v16h frag_ld(const _Float16* __restrict__ base,
                                               int ld, int r0, int k0) {
    const int L  = threadIdx.x & 31;
    const int r  = r0 + (L & 15);
    const int kh = (L >> 4) << 3;           // 0 or 8
    const _Float16* p = base + (size_t)r * ld + k0 + kh;
    v8h lo = *(const v8h*)(p);
    v8h hi = *(const v8h*)(p + 16);
    return __builtin_shufflevector(lo, hi, 0,1,2,3,4,5,6,7,8,9,10,11,12,13,14,15);
}

// ---------------- fp32 -> f16 conversion ----------------
__global__ __launch_bounds__(256)
void cvt_f32_f16(const float* __restrict__ in, _Float16* __restrict__ out, int n4) {
    int i = blockIdx.x * 256 + threadIdx.x;
    if (i >= n4) return;
    const float4 v = ((const float4*)in)[i];
    v4h o = { (_Float16)v.x, (_Float16)v.y, (_Float16)v.z, (_Float16)v.w };
    *(v4h*)(out + (size_t)i * 4) = o;
}

// ---------------- GEMM: out[m,n] = sum_k A16[m,k]*W16[n,k] + bias[n] -------
// Block tile 128x128, BK=64, double-buffered LDS fed by async loads.
// 256 threads = 8 waves; wave tile 64x32 (4x2 frags), 16 WMMA per k-step.
template<bool OUT_F16>
__global__ __launch_bounds__(256)
void gemm_wmma(const _Float16* __restrict__ A, const _Float16* __restrict__ W,
               const float* __restrict__ bias, void* __restrict__ outp,
               int M, int N, int K)
{
    constexpr int BK = 64, LDT = BK + 8;     // 72 halves/row (pad vs bank conflicts)
    __shared__ _Float16 sA[2][128 * LDT];
    __shared__ _Float16 sB[2][128 * LDT];

    const int tid  = threadIdx.x;
    const int L    = tid & 31;
    const int wave = tid >> 5;
    const int wm   = (wave >> 2) * 64;       // 0 / 64
    const int wn   = (wave & 3) * 32;        // 0..96
    const int bm   = blockIdx.y * 128;
    const int bn   = blockIdx.x * 128;

    const int row  = tid >> 1;               // 0..127
    const int col  = (tid & 1) * 32;         // halves

    auto issue_tile = [&](int buf, int k0) { // 8 async b128 per thread
        const _Float16* ga = A + (size_t)(bm + row) * K + k0 + col;
        const _Float16* gb = W + (size_t)(bn + row) * K + k0 + col;
        _Float16* la = &sA[buf][row * LDT + col];
        _Float16* lb = &sB[buf][row * LDT + col];
        #pragma unroll
        for (int c = 0; c < 4; ++c) async_ld_b128(la + c * 8, ga + c * 8);
        #pragma unroll
        for (int c = 0; c < 4; ++c) async_ld_b128(lb + c * 8, gb + c * 8);
    };

    v8f acc[4][2];
    #pragma unroll
    for (int i = 0; i < 4; ++i)
        #pragma unroll
        for (int j = 0; j < 2; ++j) acc[i][j] = v8f{};

    issue_tile(0, 0);
    for (int k0 = 0, it = 0; k0 < K; k0 += BK, ++it) {
        const int cur = it & 1;
        if (k0 + BK < K) { issue_tile(cur ^ 1, k0 + BK); async_wait8(); }
        else             { async_wait0(); }
        __syncthreads();

        #pragma unroll
        for (int kc = 0; kc < 2; ++kc) {     // two 32-deep WMMA chains
            v16h af[4], bf[2];
            #pragma unroll
            for (int i = 0; i < 4; ++i) af[i] = frag_ld(sA[cur], LDT, wm + i * 16, kc * 32);
            #pragma unroll
            for (int j = 0; j < 2; ++j) bf[j] = frag_ld(sB[cur], LDT, wn + j * 16, kc * 32);
            #pragma unroll
            for (int i = 0; i < 4; ++i)
                #pragma unroll
                for (int j = 0; j < 2; ++j)
                    acc[i][j] = __builtin_amdgcn_wmma_f32_16x16x32_f16(
                        false, af[i], false, bf[j], (short)0, acc[i][j], false, false);
        }
        __syncthreads();                     // before overwriting this buffer
    }

    // Epilogue: C layout -> lane (L&15)=N, VGPR r -> M = (L>>4)*8 + r
    const int rh = (L >> 4) << 3;
    #pragma unroll
    for (int i = 0; i < 4; ++i) {
        #pragma unroll
        for (int j = 0; j < 2; ++j) {
            const int gn = bn + wn + j * 16 + (L & 15);
            const float bv = bias ? bias[gn] : 0.f;
            #pragma unroll
            for (int r = 0; r < 8; ++r) {
                const int gm = bm + wm + i * 16 + rh + r;
                const float v = acc[i][j][r] + bv;
                if (OUT_F16) ((_Float16*)outp)[(size_t)gm * N + gn] = (_Float16)v;
                else         ((float*)outp)   [(size_t)gm * N + gn] = v;
            }
        }
    }
}

// ---------------- Flash attention: one (b,h) x 64 queries per block --------
// 128 threads = 4 waves x 16 queries; 64-key steps, double-buffered K/V.
__global__ __launch_bounds__(128)
void attn_wmma(const _Float16* __restrict__ qkv, _Float16* __restrict__ out)
{
    constexpr int LKV = HD + 8;   // 72 halves
    constexpr int LS  = 68;       // floats
    __shared__ _Float16 sK [2][64 * LKV];    // keys x hd
    __shared__ _Float16 sVT[2][64 * LKV];    // hd x keys (transposed)
    __shared__ float    sS [4][16 * LS];     // per-wave scores
    __shared__ _Float16 sP [4][16 * LKV];    // per-wave P (f16, A-layout source)
    __shared__ float    sAl[4][16];          // per-row alpha / inv-l

    const int tid  = threadIdx.x;
    const int L    = tid & 31;
    const int wave = tid >> 5;
    const int b    = blockIdx.y >> 4;
    const int h    = blockIdx.y & 15;
    const int q0   = blockIdx.x * 64;

    const _Float16* qbase = qkv + (size_t)(b * SEQ) * QKV_N + h * HD;

    // Q frags, pre-scaled by 1/sqrt(Hd)=0.125 (power of two: exact in f16)
    v16h qf[2];
    #pragma unroll
    for (int c = 0; c < 2; ++c) {
        qf[c] = frag_ld(qbase, QKV_N, q0 + wave * 16, c * 32);
        qf[c] = qf[c] * (_Float16)0.125f;
    }

    v8f of[4];
    #pragma unroll
    for (int n = 0; n < 4; ++n) of[n] = v8f{};
    float mi = -3.0e38f, li = 0.f;

    auto load_kv = [&](int buf, int kb) {    // K: 4 async b128/thread; V: transpose
        #pragma unroll
        for (int t = 0; t < 4; ++t) {
            const int c  = tid + 128 * t;    // 0..511 (8-half chunks)
            const int kr = c >> 3;           // key row 0..63
            const int c8 = (c & 7) * 8;      // hd chunk
            const size_t grow = (size_t)(b * SEQ + kb + kr) * QKV_N + h * HD + c8;
            async_ld_b128(&sK[buf][kr * LKV + c8], qkv + grow + DIMD);
            v8h d = *(const v8h*)(qkv + grow + 2 * DIMD);
            #pragma unroll
            for (int u = 0; u < 8; ++u) sVT[buf][(c8 + u) * LKV + kr] = d[u];
        }
    };

    load_kv(0, 0);
    for (int kb = 0, it = 0; kb < SEQ; kb += 64, ++it) {
        const int cur = it & 1;
        if (kb + 64 < SEQ) { load_kv(cur ^ 1, kb + 64); async_wait4(); }
        else               { async_wait0(); }
        __syncthreads();

        // S = (Q/8) K^T ; dump C frags to LDS for row-wise softmax
        const int rh = (L >> 4) << 3;
        #pragma unroll
        for (int n = 0; n < 4; ++n) {
            v8f s = v8f{};
            #pragma unroll
            for (int c = 0; c < 2; ++c) {
                v16h kf = frag_ld(sK[cur], LKV, n * 16, c * 32);
                s = __builtin_amdgcn_wmma_f32_16x16x32_f16(
                        false, qf[c], false, kf, (short)0, s, false, false);
            }
            #pragma unroll
            for (int r = 0; r < 8; ++r)
                sS[wave][(rh + r) * LS + n * 16 + (L & 15)] = s[r];
        }

        // online softmax: lanes 0..15 each own one query row
        if (L < 16) {
            float* srow = &sS[wave][L * LS];
            float mx = mi;
            #pragma unroll 8
            for (int j = 0; j < 64; ++j) mx = fmaxf(mx, srow[j]);
            const float alpha = __expf(mi - mx);
            float lnew = li * alpha;
            _Float16* prow = &sP[wave][L * LKV];
            #pragma unroll 8
            for (int j = 0; j < 64; ++j) {
                const float p = __expf(srow[j] - mx);
                lnew += p;
                prow[j] = (_Float16)p;
            }
            mi = mx; li = lnew;
            sAl[wave][L] = alpha;
        }

        // rescale O accumulators by per-row alpha
        {
            float al[8];
            #pragma unroll
            for (int r = 0; r < 8; ++r) al[r] = sAl[wave][rh + r];
            #pragma unroll
            for (int n = 0; n < 4; ++n)
                #pragma unroll
                for (int r = 0; r < 8; ++r) of[n][r] *= al[r];
        }

        // O += P * V  (P as A from sP; V^T rows give contiguous B chunks)
        #pragma unroll
        for (int c = 0; c < 2; ++c) {
            v16h pf = frag_ld(&sP[wave][0], LKV, 0, c * 32);
            #pragma unroll
            for (int n = 0; n < 4; ++n) {
                v16h vf = frag_ld(sVT[cur], LKV, n * 16, c * 32);
                of[n] = __builtin_amdgcn_wmma_f32_16x16x32_f16(
                            false, pf, false, vf, (short)0, of[n], false, false);
            }
        }
        __syncthreads();   // all waves done with buffers before next overwrite
    }

    // normalize and store: out16[b,n, h*64+hd]
    if (L < 16) sAl[wave][L] = 1.0f / li;
    const int rh = (L >> 4) << 3;
    float inv[8];
    #pragma unroll
    for (int r = 0; r < 8; ++r) inv[r] = sAl[wave][rh + r];
    #pragma unroll
    for (int n = 0; n < 4; ++n) {
        const int hd = n * 16 + (L & 15);
        #pragma unroll
        for (int r = 0; r < 8; ++r) {
            const int gq = q0 + wave * 16 + rh + r;
            out[(size_t)(b * SEQ + gq) * DIMD + h * HD + hd] = (_Float16)(of[n][r] * inv[r]);
        }
    }
}

// ---------------- host launcher ----------------
extern "C" void kernel_launch(void* const* d_in, const int* in_sizes, int n_in,
                              void* d_out, int out_size, void* d_ws, size_t ws_size,
                              hipStream_t stream)
{
    const float* x      = (const float*)d_in[0];
    const float* qkv_w  = (const float*)d_in[1];
    const float* qkv_b  = (const float*)d_in[2];
    const float* proj_w = (const float*)d_in[3];
    const float* proj_b = (const float*)d_in[4];

    // workspace layout (40 MB total):
    //   [0,8M)    x16      (reused as attention output after QKV GEMM)
    //   [8M,14M)  qkv_w16
    //   [14M,16M) proj_w16
    //   [16M,40M) qkv16
    char* ws = (char*)d_ws;
    _Float16* x16   = (_Float16*)(ws);
    _Float16* wq16  = (_Float16*)(ws + ((size_t)8  << 20));
    _Float16* wp16  = (_Float16*)(ws + ((size_t)14 << 20));
    _Float16* qkv16 = (_Float16*)(ws + ((size_t)16 << 20));

    // 1) fp32 -> f16 conversions
    {
        int n4 = MROWS * DIMD / 4;
        cvt_f32_f16<<<(n4 + 255) / 256, 256, 0, stream>>>(x, x16, n4);
        n4 = QKV_N * DIMD / 4;
        cvt_f32_f16<<<(n4 + 255) / 256, 256, 0, stream>>>(qkv_w, wq16, n4);
        n4 = DIMD * DIMD / 4;
        cvt_f32_f16<<<(n4 + 255) / 256, 256, 0, stream>>>(proj_w, wp16, n4);
    }

    // 2) fused QKV projection: [4096,1024] x [3072,1024]^T + b -> f16
    {
        dim3 grid(QKV_N / 128, MROWS / 128);   // (24, 32)
        gemm_wmma<true><<<grid, 256, 0, stream>>>(x16, wq16, qkv_b, qkv16,
                                                  MROWS, QKV_N, DIMD);
    }

    // 3) flash attention per (b,h), 64 queries per block
    {
        dim3 grid(SEQ / 64, BATCH * NHEAD);    // (32, 32)
        attn_wmma<<<grid, 128, 0, stream>>>(qkv16, /*attn out*/ x16);
    }

    // 4) output projection: [4096,1024] x [1024,1024]^T + b -> fp32 d_out
    {
        dim3 grid(DIMD / 128, MROWS / 128);    // (8, 32)
        gemm_wmma<false><<<grid, 256, 0, stream>>>(x16, wp16, proj_b, d_out,
                                                   MROWS, DIMD, DIMD);
    }
}
